// StaticDualExpertLinear_40785009442940
// MI455X (gfx1250) — matrix operand order repaired and checked by reading, not compiled, not run on hardware
//
#include <hip/hip_runtime.h>
#include <hip/hip_bf16.h>
#include <stdint.h>

// ---------------------------------------------------------------------------
// StaticDualExpertLinear for MI455X (gfx1250, wave32, WMMA)
//
//   out[m, n] = select(idx[m], x[m,:]·W_img[n,:] + b_img[n],
//                              x[m,:]·W_text[n,:] + b_text[n])
//
//   M = B*S = 8192, N = DOUT = 2048, K = DIN = 2048
//
// Strategy: split-bf16 (bf16x3) GEMM on v_wmma_f32_16x16x32_bf16, with
// double-buffered LDS stages filled by GLOBAL_LOAD_ASYNC_TO_LDS_B128
// (ASYNCcnt-tracked) when the toolchain exposes the builtin.
// ---------------------------------------------------------------------------

typedef __bf16  v16bf __attribute__((ext_vector_type(16)));
typedef __bf16  v8bf  __attribute__((ext_vector_type(8)));
typedef float   v8f   __attribute__((ext_vector_type(8)));
typedef int     v4i   __attribute__((ext_vector_type(4)));

#define M_TOT   8192    // B*S
#define N_TOT   2048    // DOUT
#define K_TOT   2048    // DIN

#define MT      128     // block tile M
#define NT      128     // block tile N
#define KT      32      // k step (one bf16 WMMA K)
#define KP      40      // padded LDS row (bf16 elems): 80B rows, 16B-aligned

#if __has_builtin(__builtin_amdgcn_global_load_async_to_lds_b128)
#define HAVE_ASYNC_LDS 1
#else
#define HAVE_ASYNC_LDS 0
#endif

// Address-space-qualified int4 pointers for the async-to-LDS builtin:
// param0 = global (AS1) int4*, param1 = LDS (AS3) int4*.
typedef __attribute__((address_space(1))) v4i glb_v4i;
typedef __attribute__((address_space(3))) v4i lds_v4i;

static __device__ inline void wait_async0() {
#if __has_builtin(__builtin_amdgcn_s_wait_asynccnt)
  __builtin_amdgcn_s_wait_asynccnt(0);
#else
  asm volatile("s_wait_asynccnt 0x0" ::: "memory");
#endif
}

// ---------------------------------------------------------------------------
// Kernel 1: split fp32 -> bf16 hi + bf16 lo (round-to-nearest-even)
// ---------------------------------------------------------------------------
__global__ __launch_bounds__(256) void split_f32_to_bf16x2(
    const float* __restrict__ src, __bf16* __restrict__ hi,
    __bf16* __restrict__ lo, int n) {
  int i = blockIdx.x * 256 + threadIdx.x;
  if (i < n) {
    float f  = src[i];
    __bf16 h = (__bf16)f;          // RNE
    float r  = f - (float)h;
    hi[i] = h;
    lo[i] = (__bf16)r;
  }
}

// ---------------------------------------------------------------------------
// LDS tile loaders: 128 x 32 bf16 tile, 16B per thread-chunk, 256 threads
// ---------------------------------------------------------------------------
static __device__ inline void stage_tile(const __bf16* __restrict__ src,
                                         int row0, int k0,
                                         __bf16 (*__restrict__ dst)[KP],
                                         int tid) {
#pragma unroll
  for (int i = tid; i < (MT * KT) / 8; i += 256) {
    const int r = i >> 2;
    const int c = (i & 3) << 3;  // 8 bf16 = 16 bytes
    const __bf16* g = src + (size_t)(row0 + r) * K_TOT + k0 + c;
#if HAVE_ASYNC_LDS
    __builtin_amdgcn_global_load_async_to_lds_b128(
        (glb_v4i*)(uintptr_t)g, (lds_v4i*)&dst[r][c], 0, 0);
#else
    const uint4 v = *reinterpret_cast<const uint4*>(g);
    *reinterpret_cast<uint4*>(&dst[r][c]) = v;
#endif
  }
}

// A fragment (16x32 bf16, M x K), per CDNA5 ISA layout:
//   lane<16 : K = {0..7, 16..23};  lane>=16 : K = {8..15, 24..31}
static __device__ inline v16bf load_a_frag(const __bf16 (*__restrict__ s)[KP],
                                           int m, int lane_hi) {
  const int k0 = lane_hi * 8;
  v8bf a = *reinterpret_cast<const v8bf*>(&s[m][k0]);
  v8bf b = *reinterpret_cast<const v8bf*>(&s[m][k0 + 16]);
  v16bf r;
#pragma unroll
  for (int i = 0; i < 8; ++i) { r[i] = a[i]; r[8 + i] = b[i]; }
  return r;
}

// B fragment (32x16 bf16, K x N), column n along a W row (W is [N][K]):
//   lane<16 : K = {0..7, 16..23};  lane>=16 : K = {8..15, 24..31}
static __device__ inline v16bf load_b_frag(const __bf16 (*__restrict__ s)[KP],
                                           int n, int lane_hi) {
  const int k0 = lane_hi * 8;
  v8bf a = *reinterpret_cast<const v8bf*>(&s[n][k0]);
  v8bf b = *reinterpret_cast<const v8bf*>(&s[n][k0 + 16]);
  v16bf r;
#pragma unroll
  for (int i = 0; i < 8; ++i) { r[i] = a[i]; r[8 + i] = b[i]; }
  return r;
}

// ---------------------------------------------------------------------------
// Kernel 2: dual-expert split-bf16 WMMA GEMM + fused bias & modality select
// grid = (N_TOT/NT, M_TOT/MT), block = 256 (8 waves: 4 along M x 2 along N)
// Double-buffered LDS (2 x 60 KB) with async-to-LDS stage fills.
// ---------------------------------------------------------------------------
__global__ __launch_bounds__(256) void dual_expert_wmma_gemm(
    const __bf16* __restrict__ xhi, const __bf16* __restrict__ xlo,
    const __bf16* __restrict__ wthi, const __bf16* __restrict__ wtlo,  // text
    const __bf16* __restrict__ wihi, const __bf16* __restrict__ wilo,  // img
    const int* __restrict__ mod_idx,
    const float* __restrict__ b_img, const float* __restrict__ b_text,
    float* __restrict__ out) {
  __shared__ __align__(16) __bf16 sA[2][2][MT][KP];     // [buf][hi/lo] 40 KB
  __shared__ __align__(16) __bf16 sB[2][2][2][NT][KP];  // [buf][x][hl] 80 KB

  const int tid     = threadIdx.x;
  const int lane    = tid & 31;
  const int wave    = tid >> 5;
  const int wave_m  = wave & 3;   // 4 waves along M, 32 rows each
  const int wave_n  = wave >> 2;  // 2 waves along N, 64 cols each
  const int lane_hi = lane >> 4;
  const int lane_lo = lane & 15;

  const int tile_m = blockIdx.y * MT;
  const int tile_n = blockIdx.x * NT;

  // acc[expert][mi][ni] : expert 0 = text, 1 = img
  v8f acc[2][2][4];
#pragma unroll
  for (int x = 0; x < 2; ++x)
#pragma unroll
    for (int mi = 0; mi < 2; ++mi)
#pragma unroll
      for (int ni = 0; ni < 4; ++ni) acc[x][mi][ni] = (v8f)(0.0f);

  auto stage = [&](int buf, int k0) {
    stage_tile(xhi,  tile_m, k0, sA[buf][0],    tid);
    stage_tile(xlo,  tile_m, k0, sA[buf][1],    tid);
    stage_tile(wthi, tile_n, k0, sB[buf][0][0], tid);
    stage_tile(wtlo, tile_n, k0, sB[buf][0][1], tid);
    stage_tile(wihi, tile_n, k0, sB[buf][1][0], tid);
    stage_tile(wilo, tile_n, k0, sB[buf][1][1], tid);
  };

  constexpr int NK = K_TOT / KT;  // 64
  stage(0, 0);

  for (int kk = 0; kk < NK; ++kk) {
    const int buf = kk & 1;
#if HAVE_ASYNC_LDS
    wait_async0();                 // my stage-(kk) async LDS writes landed
#endif
    __syncthreads();               // everyone's landed; buf^1 fully consumed
    if (kk + 1 < NK) stage(buf ^ 1, (kk + 1) * KT);

    const int mrow = wave_m * 32 + lane_lo;
    const int ncol = wave_n * 64 + lane_lo;

    v16bf aH[2], aL[2];
#pragma unroll
    for (int mi = 0; mi < 2; ++mi) {
      aH[mi] = load_a_frag(sA[buf][0], mrow + mi * 16, lane_hi);
      aL[mi] = load_a_frag(sA[buf][1], mrow + mi * 16, lane_hi);
    }

#pragma unroll
    for (int ni = 0; ni < 4; ++ni) {
#pragma unroll
      for (int x = 0; x < 2; ++x) {
        const v16bf bH = load_b_frag(sB[buf][x][0], ncol + ni * 16, lane_hi);
        const v16bf bL = load_b_frag(sB[buf][x][1], ncol + ni * 16, lane_hi);
#pragma unroll
        for (int mi = 0; mi < 2; ++mi) {
          acc[x][mi][ni] = __builtin_amdgcn_wmma_f32_16x16x32_bf16(
              false, aH[mi], false, bH, (short)0, acc[x][mi][ni], false, false);
          acc[x][mi][ni] = __builtin_amdgcn_wmma_f32_16x16x32_bf16(
              false, aH[mi], false, bL, (short)0, acc[x][mi][ni], false, false);
          acc[x][mi][ni] = __builtin_amdgcn_wmma_f32_16x16x32_bf16(
              false, aL[mi], false, bH, (short)0, acc[x][mi][ni], false, false);
        }
      }
    }
  }

  // Epilogue: bias + per-token modality select, write once.
  // C/D fragment layout: lane, vgpr v -> m = v + 8*lane_hi, n = lane&15.
#pragma unroll
  for (int ni = 0; ni < 4; ++ni) {
    const int n  = tile_n + wave_n * 64 + ni * 16 + lane_lo;
    const float bi = b_img[n];
    const float bt = b_text[n];
#pragma unroll
    for (int mi = 0; mi < 2; ++mi) {
#pragma unroll
      for (int v = 0; v < 8; ++v) {
        const int m  = tile_m + wave_m * 32 + mi * 16 + v + 8 * lane_hi;
        const int sel = mod_idx[m];
        const float r = sel ? (acc[1][mi][ni][v] + bi)
                            : (acc[0][mi][ni][v] + bt);
        out[(size_t)m * N_TOT + n] = r;
      }
    }
  }
}

// ---------------------------------------------------------------------------
extern "C" void kernel_launch(void* const* d_in, const int* in_sizes, int n_in,
                              void* d_out, int out_size, void* d_ws,
                              size_t ws_size, hipStream_t stream) {
  const float* x      = (const float*)d_in[0];
  const int*   midx   = (const int*)  d_in[1];
  const float* W_img  = (const float*)d_in[2];
  const float* b_img  = (const float*)d_in[3];
  const float* W_text = (const float*)d_in[4];
  const float* b_text = (const float*)d_in[5];
  float* out = (float*)d_out;

  // Workspace layout (bytes):
  //   xhi  @ 0        : 8192*2048*2 = 32 MiB
  //   xlo  @ 32 MiB   : 32 MiB
  //   wthi @ 64 MiB   : 8 MiB   (text hi)
  //   wtlo @ 72 MiB   : 8 MiB
  //   wihi @ 80 MiB   : 8 MiB   (img hi)
  //   wilo @ 88 MiB   : 8 MiB    -> 96 MiB total
  char* ws = (char*)d_ws;
  const size_t XN = (size_t)M_TOT * K_TOT;   // 16,777,216
  const size_t WN = (size_t)N_TOT * K_TOT;   //  4,194,304
  __bf16* xhi  = (__bf16*)(ws);
  __bf16* xlo  = (__bf16*)(ws + XN * 2);
  __bf16* wthi = (__bf16*)(ws + XN * 4);
  __bf16* wtlo = (__bf16*)(ws + XN * 4 + WN * 2);
  __bf16* wihi = (__bf16*)(ws + XN * 4 + WN * 4);
  __bf16* wilo = (__bf16*)(ws + XN * 4 + WN * 6);

  split_f32_to_bf16x2<<<(unsigned)((XN + 255) / 256), 256, 0, stream>>>(
      x, xhi, xlo, (int)XN);
  split_f32_to_bf16x2<<<(unsigned)((WN + 255) / 256), 256, 0, stream>>>(
      W_text, wthi, wtlo, (int)WN);
  split_f32_to_bf16x2<<<(unsigned)((WN + 255) / 256), 256, 0, stream>>>(
      W_img, wihi, wilo, (int)WN);

  dim3 grid(N_TOT / NT, M_TOT / MT);  // (16, 64)
  dual_expert_wmma_gemm<<<grid, 256, 0, stream>>>(
      xhi, xlo, wthi, wtlo, wihi, wilo, midx, b_img, b_text, out);
}
